// ArrowLoraLinearLayer_20959440404555
// MI455X (gfx1250) — matrix operand order repaired and checked by reading, not compiled, not run on hardware
//
#include <hip/hip_runtime.h>
#include <hip/hip_bf16.h>

typedef __attribute__((ext_vector_type(16))) _Float16 v16h;
typedef __attribute__((ext_vector_type(8)))  float    v8f;
typedef __attribute__((ext_vector_type(4)))  unsigned int uint32x4;
typedef __attribute__((ext_vector_type(8)))  int      int32x8;
typedef __attribute__((ext_vector_type(4)))  int      int32x4;

#define T_TOK 8192
#define F_DIM 2048
#define O_DIM 2048
#define N1    144    // 8 sim + 128 z, padded to multiple of 16
#define KC    64     // K chunk for GEMM1

#if defined(__has_builtin)
#  if __has_builtin(__builtin_amdgcn_tensor_load_to_lds)
#    define HAVE_TDM 1
#  else
#    define HAVE_TDM 0
#  endif
#else
#  define HAVE_TDM 0
#endif

__device__ __forceinline__ void wait_tensorcnt0() {
#if defined(__has_builtin) && __has_builtin(__builtin_amdgcn_s_wait_tensorcnt)
  __builtin_amdgcn_s_wait_tensorcnt(0);
#else
  asm volatile("s_wait_tensorcnt 0x0" ::: "memory");
#endif
}

// Forces all preceding LDS/VMEM loads to issue before anything after it:
// keeps the fragment-load batch from being sunk one-by-one into the WMMA chain.
#define SCHED_FENCE() asm volatile("" ::: "memory")

#if HAVE_TDM
// Issue one TDM load: 2D tile of tile_d0 8-byte elements per row, tile_d1 rows,
// global row stride stride_e (8-byte units), into LDS at lds_off (contiguous rows).
__device__ __forceinline__ void tdm_load(unsigned lds_off, const void* gptr,
                                         unsigned tile_d0, unsigned tile_d1,
                                         unsigned stride_e) {
  unsigned long long ga = (unsigned long long)gptr;
  uint32x4 g0 = {
      1u,                                            // count=1 (valid), user mode
      lds_off,                                       // lds_addr (bytes)
      (unsigned)(ga & 0xFFFFFFFFu),                  // global_addr[31:0]
      (unsigned)((ga >> 32) & 0x1FFFFFFu) | 0x80000000u  // addr[56:32] | type=2
  };
  const unsigned td = 0x40000000u;                   // huge tensor dims: no OOB clip
  int32x8 g1;
  g1[0] = 0x30000;                                   // data_size = 3 (8 bytes)
  g1[1] = (int)((td & 0xFFFFu) << 16);               // tensor_dim0[15:0]
  g1[2] = (int)(((td >> 16) & 0xFFFFu) |             // tensor_dim0[31:16]
                ((td & 0xFFFFu) << 16));             // tensor_dim1[15:0]
  g1[3] = (int)(((td >> 16) & 0xFFFFu) |             // tensor_dim1[31:16]
                ((tile_d0 & 0xFFFFu) << 16));        // tile_dim0
  g1[4] = (int)(tile_d1 & 0xFFFFu);                  // tile_dim1 (tile_dim2 = 0)
  g1[5] = (int)stride_e;                             // tensor_dim0_stride[31:0]
  g1[6] = 0;                                         // stride[47:32] | dim1_stride lo
  g1[7] = 0;
  int32x4 z4 = {0, 0, 0, 0};
#if __clang_major__ >= 23
  int32x8 z8 = {0, 0, 0, 0, 0, 0, 0, 0};
  __builtin_amdgcn_tensor_load_to_lds(g0, g1, z4, z4, z8, 0);
#else
  __builtin_amdgcn_tensor_load_to_lds(g0, g1, z4, z4, 0);
#endif
}
#endif

// ---------------- prep: pack protos + A_stack into f16 W1 [144][2048] -------
__global__ __launch_bounds__(256) void prep_w1_kernel(
    const float* __restrict__ protos, const float* __restrict__ A,
    _Float16* __restrict__ W1h) {
  int idx = blockIdx.x * 256 + threadIdx.x;           // 144*2048 total
  int n = idx >> 11, k = idx & 2047;
  float v = 0.0f;
  if (n < 8)        v = protos[n * F_DIM + k];
  else if (n < 136) v = A[(n - 8) * F_DIM + k];       // A_stack flat [E*r][F]
  W1h[idx] = (_Float16)v;
}

// ---------------- prep: transpose B_stack [E,O,r] -> f16 B2t [O][128] -------
__global__ __launch_bounds__(256) void prep_b2_kernel(
    const float* __restrict__ Bst, _Float16* __restrict__ B2t) {
  int idx = blockIdx.x * 256 + threadIdx.x;           // 2048*128 total
  int o = idx >> 7, kp = idx & 127;                   // kp = e*16 + j
  int e = kp >> 4, j = kp & 15;
  B2t[idx] = (_Float16)Bst[(e * O_DIM + o) * 16 + j];
}

// ---- GEMM1 (K-split): C1p[part][8192][144] = x[:, kpart] @ W1[:, kpart]^T --
__global__ __launch_bounds__(256) void gemm1_kernel(
    const float* __restrict__ x, const _Float16* __restrict__ W1h,
    float* __restrict__ C1p, int kLen) {
  __shared__ __align__(16) _Float16 xs[128 * KC];
  __shared__ __align__(16) _Float16 ws[N1 * KC];
  const int tid  = threadIdx.x;
  const int w    = tid >> 5;
  const int lane = tid & 31;
  const int Mbase  = blockIdx.x * 128;
  const int part   = blockIdx.y;
  const int kBegin = part * kLen;
  const int kEnd   = kBegin + kLen;
  float* __restrict__ C1 = C1p + (size_t)part * (T_TOK * N1);

  v8f acc[9] = {};

  for (int k0 = kBegin; k0 < kEnd; k0 += KC) {
#if HAVE_TDM
    if (w == 0) {   // wave-level DMA: W1 tile 144 rows x 128B, row stride 4KB
      tdm_load((unsigned)(unsigned long long)(void*)ws,
               W1h + k0, KC * 2 / 8, N1, F_DIM * 2 / 8);
    }
#endif
    // stage x tile (128 x 64 f32 -> f16), coalesced float4 loads (overlaps TDM)
    #pragma unroll
    for (int it = 0; it < 8; ++it) {
      int idx = tid + it * 256;                 // float4 index, 128*16 total
      int row = idx >> 4, c4 = idx & 15;
      const float* src = x + (Mbase + row) * F_DIM + k0 + c4 * 4;
      const float4 v = *(const float4*)src;
      if (k0 + KC < kEnd) __builtin_prefetch(src + KC, 0, 0);
      _Float16* dst = &xs[row * KC + c4 * 4];
      dst[0] = (_Float16)v.x; dst[1] = (_Float16)v.y;
      dst[2] = (_Float16)v.z; dst[3] = (_Float16)v.w;
    }
#if HAVE_TDM
    if (w == 0) wait_tensorcnt0();
#else
    for (int idx = tid; idx < (N1 * KC) / 8; idx += 256) {
      int row = idx >> 3, c4 = idx & 7;
      *(uint4*)&ws[row * KC + c4 * 8] =
          *(const uint4*)(W1h + row * F_DIM + k0 + c4 * 8);
    }
#endif
    __syncthreads();

    #pragma unroll
    for (int kk = 0; kk < KC; kk += 32) {
      // A fragment: 16x32 f16 per ISA layout (two 16B runs)
      const int arow  = w * 16 + (lane & 15);
      const int abase = kk + ((lane >> 4) << 3);
      v16h a;
      *((uint4*)&a)     = *(const uint4*)&xs[arow * KC + abase];
      *((uint4*)&a + 1) = *(const uint4*)&xs[arow * KC + abase + 16];
      // batch all B fragments, fence, then WMMA chain
      const int bbase = kk + ((lane >> 4) << 3);
      v16h bf[9];
      #pragma unroll
      for (int nt = 0; nt < 9; ++nt) {
        const int ncol = nt * 16 + (lane & 15);
        *((uint4*)&bf[nt])     = *(const uint4*)&ws[ncol * KC + bbase];
        *((uint4*)&bf[nt] + 1) = *(const uint4*)&ws[ncol * KC + bbase + 16];
      }
      SCHED_FENCE();
      #pragma unroll
      for (int nt = 0; nt < 9; ++nt)
        acc[nt] = __builtin_amdgcn_wmma_f32_16x16x32_f16(
            false, a, false, bf[nt], (short)0, acc[nt], false, false);
    }
    __syncthreads();
  }

  const int rbase = Mbase + w * 16 + ((lane >> 4) << 3);
  const int cbase = lane & 15;
  #pragma unroll
  for (int nt = 0; nt < 9; ++nt)
    #pragma unroll
    for (int v = 0; v < 8; ++v)
      C1[(rbase + v) * N1 + nt * 16 + cbase] = acc[nt][v];
}

// -------- routing: sum K-partials, |sim| top-2 softmax * scales -> f16 z ----
__global__ __launch_bounds__(256) void route_kernel(
    const float* __restrict__ C1p, const float* __restrict__ scales,
    _Float16* __restrict__ ZWh, int nparts) {
  const int t = blockIdx.x * 256 + threadIdx.x;       // 8192 tokens
  float acc[N1];
  #pragma unroll
  for (int i = 0; i < N1; ++i) acc[i] = 0.0f;
  for (int p = 0; p < nparts; ++p) {                  // fixed order: deterministic
    const float* row = C1p + (size_t)p * (T_TOK * N1) + t * N1;
    #pragma unroll
    for (int i = 0; i < N1 / 4; ++i) {
      const float4 v = *(const float4*)(row + i * 4);
      acc[4 * i + 0] += v.x; acc[4 * i + 1] += v.y;
      acc[4 * i + 2] += v.z; acc[4 * i + 3] += v.w;
    }
  }
  float s[8];
  #pragma unroll
  for (int e = 0; e < 8; ++e) s[e] = fabsf(acc[e]);
  int i0 = 0;
  #pragma unroll
  for (int e = 1; e < 8; ++e) if (s[e] > s[i0]) i0 = e;   // first max (jax tie rule)
  const float v0 = s[i0];
  int i1 = 0; float v1 = -1.0f;                           // sims >= 0
  #pragma unroll
  for (int e = 0; e < 8; ++e)
    if (e != i0 && s[e] > v1) { v1 = s[e]; i1 = e; }
  const float e1  = __expf(v1 - v0);
  const float inv = 1.0f / (1.0f + e1);
  float cw[8];
  #pragma unroll
  for (int e = 0; e < 8; ++e) cw[e] = 0.0f;
  cw[i0] = scales[i0] * inv;
  cw[i1] = scales[i1] * e1 * inv;

  _Float16 zr[128];
  #pragma unroll
  for (int e = 0; e < 8; ++e) {
    const float wgt = cw[e];
    #pragma unroll
    for (int j = 0; j < 16; ++j)
      zr[e * 16 + j] = (_Float16)(acc[8 + e * 16 + j] * wgt);
  }
  uint4* dst = (uint4*)(ZWh + t * 128);
  const uint4* src = (const uint4*)zr;
  #pragma unroll
  for (int i = 0; i < 16; ++i) dst[i] = src[i];
}

// ---------------- GEMM2: out[8192][2048] = zw[8192][128] @ B2t^T ------------
__global__ __launch_bounds__(256) void gemm2_kernel(
    const _Float16* __restrict__ ZWh, const _Float16* __restrict__ B2t,
    float* __restrict__ out) {
  __shared__ __align__(16) _Float16 zs[128 * 128];
  __shared__ __align__(16) _Float16 bs[128 * 128];
  const int tid  = threadIdx.x;
  const int w    = tid >> 5;
  const int lane = tid & 31;
  const int Mbase = blockIdx.x * 128;
  const int Nbase = blockIdx.y * 128;

#if HAVE_TDM
  if (w == 0) {   // both tiles are contiguous 32KB blocks -> 1D TDM copies
    tdm_load((unsigned)(unsigned long long)(void*)zs,
             ZWh + Mbase * 128, 4096, 1, 4096);
    tdm_load((unsigned)(unsigned long long)(void*)bs,
             B2t + Nbase * 128, 4096, 1, 4096);
    wait_tensorcnt0();
  }
#else
  #pragma unroll
  for (int it = 0; it < 8; ++it) {
    int idx = tid + it * 256;                 // uint4 index, 128*16 total
    int row = idx >> 4, c4 = idx & 15;
    *(uint4*)&zs[row * 128 + c4 * 8] =
        *(const uint4*)(ZWh + (Mbase + row) * 128 + c4 * 8);
    *(uint4*)&bs[row * 128 + c4 * 8] =
        *(const uint4*)(B2t + (Nbase + row) * 128 + c4 * 8);
  }
#endif
  __syncthreads();

  v8f acc[8] = {};
  #pragma unroll
  for (int kk = 0; kk < 128; kk += 32) {
    const int arow  = w * 16 + (lane & 15);
    const int abase = kk + ((lane >> 4) << 3);
    v16h a;
    *((uint4*)&a)     = *(const uint4*)&zs[arow * 128 + abase];
    *((uint4*)&a + 1) = *(const uint4*)&zs[arow * 128 + abase + 16];
    const int bbase = kk + ((lane >> 4) << 3);
    v16h bf[8];
    #pragma unroll
    for (int nt = 0; nt < 8; ++nt) {
      const int ncol = nt * 16 + (lane & 15);
      *((uint4*)&bf[nt])     = *(const uint4*)&bs[ncol * 128 + bbase];
      *((uint4*)&bf[nt] + 1) = *(const uint4*)&bs[ncol * 128 + bbase + 16];
    }
    SCHED_FENCE();
    #pragma unroll
    for (int nt = 0; nt < 8; ++nt)
      acc[nt] = __builtin_amdgcn_wmma_f32_16x16x32_f16(
          false, a, false, bf[nt], (short)0, acc[nt], false, false);
  }

  const int rbase = Mbase + w * 16 + ((lane >> 4) << 3);
  const int cbase = lane & 15;
  #pragma unroll
  for (int nt = 0; nt < 8; ++nt)
    #pragma unroll
    for (int v = 0; v < 8; ++v)
      out[(rbase + v) * O_DIM + Nbase + nt * 16 + cbase] = acc[nt][v];
}

// ---------------------------------------------------------------------------
extern "C" void kernel_launch(void* const* d_in, const int* in_sizes, int n_in,
                              void* d_out, int out_size, void* d_ws, size_t ws_size,
                              hipStream_t stream) {
  const float* x      = (const float*)d_in[0];   // [4,2048,2048]
  const float* protos = (const float*)d_in[1];   // [8,2048]
  const float* A      = (const float*)d_in[2];   // [8,16,2048]
  const float* Bst    = (const float*)d_in[3];   // [8,2048,16]
  const float* scales = (const float*)d_in[4];   // [8]
  // d_in[5] = top_k (== 2, specialized)

  char* ws = (char*)d_ws;
  const size_t C1_OFF = 1114112;                         // after W1h + B2t
  const size_t C1_SZ  = (size_t)T_TOK * N1 * 4;          // 4,718,592 B per part
  const size_t ZW_SZ  = (size_t)T_TOK * 128 * 2;         // 2,097,152 B

  // K-split factor: 4 if workspace allows (4x occupancy for GEMM1), else 1.
  int ksplit = (ws_size >= C1_OFF + 4 * C1_SZ + ZW_SZ) ? 4 : 1;

  _Float16* W1h = (_Float16*)(ws);                       //   589,824 B
  _Float16* B2t = (_Float16*)(ws + 589824);              //   524,288 B
  float*    C1p = (float*)   (ws + C1_OFF);
  _Float16* ZWh = (_Float16*)(ws + C1_OFF + (size_t)ksplit * C1_SZ);
  float* out = (float*)d_out;

  prep_w1_kernel<<<(N1 * F_DIM) / 256, 256, 0, stream>>>(protos, A, W1h);
  prep_b2_kernel<<<(O_DIM * 128) / 256, 256, 0, stream>>>(Bst, B2t);
  gemm1_kernel<<<dim3(T_TOK / 128, ksplit), 256, 0, stream>>>(
      x, W1h, C1p, F_DIM / ksplit);
  route_kernel<<<T_TOK / 256, 256, 0, stream>>>(C1p, scales, ZWh, ksplit);
  gemm2_kernel<<<dim3(T_TOK / 128, O_DIM / 128), 256, 0, stream>>>(ZWh, B2t, out);
}
